// MultiHeadAttention_4947802325526
// MI455X (gfx1250) — matrix operand order repaired and checked by reading, not compiled, not run on hardware
//
#include <hip/hip_runtime.h>
#include <math.h>

typedef __attribute__((ext_vector_type(16))) __bf16 v16bf;
typedef __attribute__((ext_vector_type(8)))  float  v8f;

#define DEV __device__ __forceinline__

namespace {

constexpr int B_  = 4;
constexpr int S_  = 4096;
constexpr int E_  = 512;
constexpr int H_  = 8;
constexpr int Dh_ = 64;
constexpr int NQKV_ = 3 * E_;     // 1536
constexpr int M_  = B_ * S_;      // 16384

union PairBF { unsigned u; __bf16 h[2]; };

// A fragment (16x32 bf16, MxK). ISA 7.12.2: lane<16 holds M=lane,
// VGPR0-3: K = half*8 + {0..7}, VGPR4-7: K = 16 + half*8 + {0..7}, packed pairs.
DEV v16bf load_frag_a(const __bf16* base, int stride) {
  const int lane = threadIdx.x & 31;
  const int hf = lane >> 4;
  const int m  = lane & 15;
  const __bf16* row = base + m * stride;
  union { v16bf v; unsigned u[8]; } f;
#pragma unroll
  for (int i = 0; i < 4; ++i)
    f.u[i] = *(const unsigned*)(row + hf * 8 + 2 * i);
#pragma unroll
  for (int i = 0; i < 4; ++i)
    f.u[4 + i] = *(const unsigned*)(row + 16 + hf * 8 + 2 * i);
  return f.v;
}

// B fragment (32x16 bf16, KxN), sourced from LDS laid out [N][K-contiguous].
// Same K order as A: VGPR0-3: K = half*8 + {0..7}, VGPR4-7: K = 16 + half*8 + {0..7}.
DEV v16bf load_frag_b(const __bf16* base, int stride) {
  const int lane = threadIdx.x & 31;
  const int hf = lane >> 4;
  const int n  = lane & 15;
  const __bf16* row = base + n * stride;
  union { v16bf v; unsigned u[8]; } f;
#pragma unroll
  for (int i = 0; i < 4; ++i)
    f.u[i] = *(const unsigned*)(row + hf * 8 + 2 * i);
#pragma unroll
  for (int i = 0; i < 4; ++i)
    f.u[4 + i] = *(const unsigned*)(row + 16 + hf * 8 + 2 * i);
  return f.v;
}

DEV v8f wmma_bf16(v16bf a, v16bf b, v8f c) {
  return __builtin_amdgcn_wmma_f32_16x16x32_bf16(
      false, a, false, b, (short)0, c, false, false);
}

// Async global->LDS DWORD copy (CDNA5 ASYNCcnt path, §15.18 opcode 96).
// ldsaddr: LDS byte address (generic-pointer low 32 bits == LDS offset).
DEV void async_g2l_b32(unsigned ldsaddr, const unsigned* gaddr) {
  asm volatile("global_load_async_to_lds_b32 %0, %1, off"
               :: "v"(ldsaddr), "v"(gaddr) : "memory");
}

DEV void wait_asynccnt0() {
  asm volatile("s_wait_asynccnt 0x0" ::: "memory");
}

} // namespace

// ---------------------------------------------------------------------------
// Kernel 1: QKV projection. C[M,1536] = x[M,512] @ Wqkv + bqkv, scattered to
// bf16 workspace laid out [3][B,H,S,Dh].
// ---------------------------------------------------------------------------
__global__ __launch_bounds__(256) void qkv_gemm_kernel(
    const float* __restrict__ x, const float* __restrict__ Wqkv,
    const float* __restrict__ bqkv, __bf16* __restrict__ qkv_ws) {
  __shared__ __bf16 As[128 * 34];   // [m][k], stride 34 (odd-DWORD, 4B aligned rows)
  __shared__ __bf16 Bs[128 * 34];   // [n][k]

  const int tid = threadIdx.x;
  const int n0 = blockIdx.x * 128;
  const int m0 = blockIdx.y * 128;
  const int wid = tid >> 5;
  const int wm = wid >> 2;          // 0..1 -> 64 rows
  const int wn = wid & 3;           // 0..3 -> 32 cols
  const int lane = tid & 31;
  const int hf = lane >> 4;
  const int nl = lane & 15;

  v8f acc[4][2] = {};

  for (int kt = 0; kt < E_; kt += 32) {
    __syncthreads();
    if (kt + 32 < E_)
      __builtin_prefetch(x + (size_t)(m0 + (tid >> 1)) * E_ + kt + 32, 0, 0);
#pragma unroll
    for (int i = 0; i < 16; ++i) {        // A: 128x32 f32 -> bf16
      int e = tid + i * 256;
      int r = e >> 5, k = e & 31;
      As[r * 34 + k] = (__bf16)x[(size_t)(m0 + r) * E_ + kt + k];
    }
#pragma unroll
    for (int i = 0; i < 16; ++i) {        // B: transpose 32x128 -> [n][k]
      int e = tid + i * 256;
      int k = e >> 7, n = e & 127;
      Bs[n * 34 + k] = (__bf16)Wqkv[(size_t)(kt + k) * NQKV_ + n0 + n];
    }
    __syncthreads();

    v16bf bfrag[2];
#pragma unroll
    for (int j = 0; j < 2; ++j)
      bfrag[j] = load_frag_b(Bs + (wn * 32 + j * 16) * 34, 34);
#pragma unroll
    for (int i = 0; i < 4; ++i) {
      v16bf afrag = load_frag_a(As + (wm * 64 + i * 16) * 34, 34);
#pragma unroll
      for (int j = 0; j < 2; ++j)
        acc[i][j] = wmma_bf16(afrag, bfrag[j], acc[i][j]);
    }
  }

#pragma unroll
  for (int i = 0; i < 4; ++i)
#pragma unroll
    for (int j = 0; j < 2; ++j)
#pragma unroll
      for (int r = 0; r < 8; ++r) {
        int m = m0 + wm * 64 + i * 16 + hf * 8 + r;
        int n = n0 + wn * 32 + j * 16 + nl;
        float v = acc[i][j][r] + bqkv[n];
        int which = n >> 9, rem = n & 511;
        int hh = rem >> 6, dd = rem & 63;
        int bb = m >> 12, ss = m & (S_ - 1);
        size_t idx = (((size_t)(which * B_ + bb) * H_ + hh) * S_ + ss) * Dh_ + dd;
        qkv_ws[idx] = (__bf16)v;
      }
}

// ---------------------------------------------------------------------------
// Kernel 2: causal flash attention. Grid (S/64, B*H); 4 waves x 16 q-rows.
// Q/K tiles staged via GLOBAL_LOAD_ASYNC_TO_LDS (no VGPR round-trip);
// V staged through VGPRs (transpose).
// ---------------------------------------------------------------------------
__global__ __launch_bounds__(128) void flash_attn_kernel(
    const __bf16* __restrict__ qkv_ws, __bf16* __restrict__ attn_ws) {
  __shared__ __bf16 Qs[64 * 66];    // [q-row][d]
  __shared__ __bf16 Ks[32 * 66];    // [t][d]  == B-fragment layout for Q@K^T
  __shared__ __bf16 Vts[64 * 34];   // [d][t]  == B-fragment layout for P@V
  __shared__ __bf16 Ps[64 * 34];    // per-wave 16 rows of P, [m][t]

  const int tid  = threadIdx.x;
  const int wid  = tid >> 5;
  const int lane = tid & 31;
  const int hf   = lane >> 4;
  const int nl   = lane & 15;

  const int qb = blockIdx.x;
  const int bh = blockIdx.y;
  const size_t chunk = (size_t)B_ * H_ * S_ * Dh_;
  const __bf16* qp = qkv_ws + (size_t)bh * S_ * Dh_;
  const __bf16* kp = qp + chunk;
  const __bf16* vp = qp + 2 * chunk;

  const unsigned qs_lds = (unsigned)(size_t)(void*)Qs;  // LDS byte address
  const unsigned ks_lds = (unsigned)(size_t)(void*)Ks;

  {  // async-stage Q tile: 64 rows x 32 dwords, rows qb*64..+63
    const unsigned* qsrc = (const unsigned*)qp;
#pragma unroll
    for (int i = 0; i < 16; ++i) {
      int e = tid + i * 128;
      int r = e >> 5, dp = e & 31;
      async_g2l_b32(qs_lds + (unsigned)(r * 132 + dp * 4),
                    qsrc + (size_t)(qb * 64 + r) * 32 + dp);
    }
    wait_asynccnt0();
  }
  __syncthreads();

  v16bf qfrag[2];
#pragma unroll
  for (int ks = 0; ks < 2; ++ks)
    qfrag[ks] = load_frag_a(Qs + (wid * 16) * 66 + ks * 32, 66);

  v8f acc_o[4] = {};
  float m_run[8], l_run[8];
#pragma unroll
  for (int r = 0; r < 8; ++r) { m_run[r] = -INFINITY; l_run[r] = 0.f; }

  const float scale = 0.125f;  // 1/sqrt(Dh)
  const int row_base = qb * 64 + wid * 16 + hf * 8;
  const int t_end = (qb + 1) * 64;

  for (int t0 = 0; t0 < t_end; t0 += 32) {
    __syncthreads();
    {   // async-stage K [t][d]; stage V transposed [d][t] through VGPRs
      const unsigned* ksrc = (const unsigned*)kp;
#pragma unroll
      for (int i = 0; i < 8; ++i) {
        int e = tid + i * 128;
        int t = e >> 5, dp = e & 31;
        async_g2l_b32(ks_lds + (unsigned)(t * 132 + dp * 4),
                      ksrc + (size_t)(t0 + t) * 32 + dp);
      }
      const unsigned* vsrc = (const unsigned*)vp;
#pragma unroll
      for (int i = 0; i < 8; ++i) {
        int e = tid + i * 128;
        int t = e >> 5, dp = e & 31;
        PairBF pw; pw.u = vsrc[(size_t)(t0 + t) * 32 + dp];
        Vts[(2 * dp) * 34 + t]     = pw.h[0];
        Vts[(2 * dp + 1) * 34 + t] = pw.h[1];
      }
      wait_asynccnt0();
    }
    __syncthreads();

    // S = Q @ K^T : 16x32 scores per wave (two 16x16 tiles, K-dim 64 = 2 steps)
    v8f s0 = {}, s1 = {};
#pragma unroll
    for (int ks = 0; ks < 2; ++ks) {
      v16bf kb0 = load_frag_b(Ks + 0 * 66 + ks * 32, 66);
      v16bf kb1 = load_frag_b(Ks + 16 * 66 + ks * 32, 66);
      s0 = wmma_bf16(qfrag[ks], kb0, s0);
      s1 = wmma_bf16(qfrag[ks], kb1, s1);
    }

    // online softmax; row r lives in lanes of this half (shfl stays in-half)
#pragma unroll
    for (int r = 0; r < 8; ++r) {
      const int row = row_base + r;
      float v0 = (t0 + nl      <= row) ? s0[r] * scale : -INFINITY;
      float v1 = (t0 + 16 + nl <= row) ? s1[r] * scale : -INFINITY;
      float mx = fmaxf(v0, v1);
#pragma unroll
      for (int off = 8; off >= 1; off >>= 1)
        mx = fmaxf(mx, __shfl_xor(mx, off, 32));
      float mnew = fmaxf(m_run[r], mx);
      float corr = __expf(m_run[r] - mnew);
      float p0 = __expf(v0 - mnew);
      float p1 = __expf(v1 - mnew);
      float sum = p0 + p1;
#pragma unroll
      for (int off = 8; off >= 1; off >>= 1)
        sum += __shfl_xor(sum, off, 32);
      l_run[r] = l_run[r] * corr + sum;
      m_run[r] = mnew;
      __bf16* prow = Ps + (wid * 16 + hf * 8 + r) * 34;
      prow[nl]      = (__bf16)p0;
      prow[16 + nl] = (__bf16)p1;
#pragma unroll
      for (int dj = 0; dj < 4; ++dj) acc_o[dj][r] *= corr;
    }
    __syncthreads();  // make P / Vts visible for fragment gathers

    v16bf pfrag = load_frag_a(Ps + (wid * 16) * 34, 34);
#pragma unroll
    for (int dj = 0; dj < 4; ++dj) {
      v16bf vfrag = load_frag_b(Vts + (dj * 16) * 34, 34);
      acc_o[dj] = wmma_bf16(pfrag, vfrag, acc_o[dj]);
    }
  }

  const int b = bh >> 3, h = bh & 7;
#pragma unroll
  for (int r = 0; r < 8; ++r) {
    float inv = 1.f / l_run[r];
    int qs = qb * 64 + wid * 16 + hf * 8 + r;
    size_t base = ((size_t)b * S_ + qs) * E_ + h * Dh_;
#pragma unroll
    for (int dj = 0; dj < 4; ++dj)
      attn_ws[base + dj * 16 + nl] = (__bf16)(acc_o[dj][r] * inv);
  }
}

// ---------------------------------------------------------------------------
// Kernel 3: output projection. out[M,512] = attn[M,512] @ Wo + bo (fp32 out).
// ---------------------------------------------------------------------------
__global__ __launch_bounds__(256) void out_gemm_kernel(
    const __bf16* __restrict__ attn, const float* __restrict__ Wo,
    const float* __restrict__ bo, float* __restrict__ out) {
  __shared__ __bf16 As[128 * 34];
  __shared__ __bf16 Bs[128 * 34];

  const int tid = threadIdx.x;
  const int n0 = blockIdx.x * 128;
  const int m0 = blockIdx.y * 128;
  const int wid = tid >> 5;
  const int wm = wid >> 2, wn = wid & 3;
  const int lane = tid & 31, hf = lane >> 4, nl = lane & 15;

  v8f acc[4][2] = {};

  for (int kt = 0; kt < E_; kt += 32) {
    __syncthreads();
    if (kt + 32 < E_)
      __builtin_prefetch(attn + (size_t)(m0 + (tid >> 1)) * E_ + kt + 32, 0, 0);
    {
      const unsigned* asrc = (const unsigned*)attn;
      unsigned* adst = (unsigned*)As;   // row stride 17 dwords
#pragma unroll
      for (int i = 0; i < 8; ++i) {
        int e = tid + i * 256;
        int r = e >> 4, kpair = e & 15;
        adst[r * 17 + kpair] = asrc[(size_t)(m0 + r) * (E_ / 2) + (kt >> 1) + kpair];
      }
    }
#pragma unroll
    for (int i = 0; i < 16; ++i) {
      int e = tid + i * 256;
      int k = e >> 7, n = e & 127;
      Bs[n * 34 + k] = (__bf16)Wo[(size_t)(kt + k) * E_ + n0 + n];
    }
    __syncthreads();

    v16bf bfrag[2];
#pragma unroll
    for (int j = 0; j < 2; ++j)
      bfrag[j] = load_frag_b(Bs + (wn * 32 + j * 16) * 34, 34);
#pragma unroll
    for (int i = 0; i < 4; ++i) {
      v16bf afrag = load_frag_a(As + (wm * 64 + i * 16) * 34, 34);
#pragma unroll
      for (int j = 0; j < 2; ++j)
        acc[i][j] = wmma_bf16(afrag, bfrag[j], acc[i][j]);
    }
  }

#pragma unroll
  for (int i = 0; i < 4; ++i)
#pragma unroll
    for (int j = 0; j < 2; ++j)
#pragma unroll
      for (int r = 0; r < 8; ++r) {
        int m = m0 + wm * 64 + i * 16 + hf * 8 + r;
        int n = n0 + wn * 32 + j * 16 + nl;
        out[(size_t)m * E_ + n] = acc[i][j][r] + bo[n];
      }
}

// ---------------------------------------------------------------------------
extern "C" void kernel_launch(void* const* d_in, const int* in_sizes, int n_in,
                              void* d_out, int out_size, void* d_ws, size_t ws_size,
                              hipStream_t stream) {
  const float* x    = (const float*)d_in[0];
  const float* Wqkv = (const float*)d_in[1];
  const float* bqkv = (const float*)d_in[2];
  const float* Wo   = (const float*)d_in[3];
  const float* bo   = (const float*)d_in[4];
  float* out = (float*)d_out;

  const size_t chunk = (size_t)B_ * H_ * S_ * Dh_;       // 8,388,608 elems
  __bf16* qkv_ws  = (__bf16*)d_ws;                       // 3*chunk bf16 (~50 MB)
  __bf16* attn_ws = qkv_ws + 3 * chunk;                  // M*E bf16 (~17 MB)

  qkv_gemm_kernel<<<dim3(NQKV_ / 128, M_ / 128), 256, 0, stream>>>(
      x, Wqkv, bqkv, qkv_ws);
  flash_attn_kernel<<<dim3(S_ / 64, B_ * H_), 128, 0, stream>>>(
      qkv_ws, attn_ws);
  out_gemm_kernel<<<dim3(E_ / 128, M_ / 128), 256, 0, stream>>>(
      attn_ws, Wo, bo, out);
}
